// ADNN_57501022159565
// MI455X (gfx1250) — compile-verified, matrix operand order checked by reading, not run on hardware
//
#include <hip/hip_runtime.h>
#include <hip/hip_bf16.h>

typedef __bf16 bf16;
typedef __attribute__((ext_vector_type(16))) __bf16 v16bf;
typedef __attribute__((ext_vector_type(4)))  __bf16 v4bf;
typedef __attribute__((ext_vector_type(8)))  float  v8f;
typedef __attribute__((ext_vector_type(4)))  float  v4f_t;

// TDM descriptor operand types (clang-23 / therock 6-arg builtin:
//  (uint32x4 g0, int32x8 g1, int32x4 g2, int32x4 g3, int32x8 ext, i32 cpol))
typedef __attribute__((ext_vector_type(4))) unsigned tdm_g0_t;
typedef __attribute__((ext_vector_type(8))) int      tdm_g1_t;
typedef __attribute__((ext_vector_type(4))) int      tdm_g2_t;
typedef __attribute__((ext_vector_type(8))) int      tdm_g4_t;

#define BM 128
#define BN 64
#define BK 32

enum { EPI_BIAS_DUAL = 0, EPI_PLAIN = 1, EPI_ADD_RELU = 2, EPI_BIAS_F32 = 3 };

__device__ __forceinline__ uint2 pack4_bf16(float4 f) {
    v4f_t v; v.x = f.x; v.y = f.y; v.z = f.z; v.w = f.w;
    v4bf b = __builtin_convertvector(v, v4bf);   // 2x v_cvt_pk_bf16_f32
    union { v4bf b; uint2 u; } p; p.b = b;
    return p.u;
}

// ---------------------------------------------------------------------------
// Tensor Data Mover: 2D bf16 tile (tile0 x tile1 rows) global -> LDS.
// D# per CDNA5 ISA ch.8: group0 = {count, lds_addr, global_addr, type=2},
// group1 = {data_size=2B, tensor_dim0/1, tile_dim0/1, dim0 stride}.
// Groups 2/3 + extended group zero (2D tensor). Issue from ONE wave only;
// completion tracked with TENSORcnt.
// ---------------------------------------------------------------------------
__device__ __forceinline__ void tdm_load_tile_bf16(
    unsigned ldsAddr, const void* gptr,
    unsigned dim0, unsigned dim1, unsigned stride0,
    unsigned tile0, unsigned tile1) {
    unsigned long long ga = (unsigned long long)(uintptr_t)gptr;
    tdm_g0_t g0 = { 1u,                                   // count=1 (valid D#)
                    ldsAddr,                              // LDS byte address
                    (unsigned)ga,                         // global addr [31:0]
                    (unsigned)((ga >> 32) & 0x01ffffffull) | 0x80000000u }; // [56:32] | type=2
    tdm_g1_t g1 = { (int)0x00010000u,                     // data_size=1 -> 2 bytes
                    (int)((dim0 & 0xffffu) << 16),        // tensor_dim0[15:0]
                    (int)((dim0 >> 16) | ((dim1 & 0xffffu) << 16)),
                    (int)((dim1 >> 16) | ((tile0 & 0xffffu) << 16)),
                    (int)(tile1 & 0xffffu),               // tile_dim1 (tile_dim2=0)
                    (int)stride0,                         // tensor_dim0_stride lo32
                    0, 0 };
    tdm_g2_t gz4 = { 0, 0, 0, 0 };
    tdm_g4_t gz8 = { 0, 0, 0, 0, 0, 0, 0, 0 };
    __builtin_amdgcn_tensor_load_to_lds(g0, g1, gz4, gz4, gz8, 0);
}

// ---------------------------------------------------------------------------
// Row-wise softmax for the 128x128 attention matrix (tiny; one block).
// ---------------------------------------------------------------------------
__global__ void softmax_rows_kernel(const float* __restrict__ att,
                                    float* __restrict__ conn) {
    const int r = threadIdx.x;            // 128 threads, one row each
    const float* row = att + r * 128;
    float m = -3.4e38f;
    for (int c = 0; c < 128; ++c) m = fmaxf(m, row[c]);
    float s = 0.f;
    for (int c = 0; c < 128; ++c) s += __expf(row[c] - m);
    const float inv = 1.0f / s;
    float* orow = conn + r * 128;
    for (int c = 0; c < 128; ++c) orow[c] = __expf(row[c] - m) * inv;
}

// ---------------------------------------------------------------------------
// bf16 WMMA GEMM. A-source bf16 -> TDM (async DMA, double-buffered LDS).
// A-source f32 -> VGPR staging + packed cvt. B staged via VGPRs, stored
// transposed [n][k]. Block tile 128x64, 8 waves x (16 x 64), K-step 32.
// ---------------------------------------------------------------------------
template <bool ABF, bool BBF, int EPI>
__global__ __launch_bounds__(256)
void gemm_wmma_kernel(const void* __restrict__ Abase, long aBatch, int lda,
                      const void* __restrict__ Bbase, long bBatch, int ldb,
                      int K,
                      const float* __restrict__ bias, long biasBatch,
                      const bf16* __restrict__ addend, long addBatch,
                      void* __restrict__ outPrim, long outBatch,
                      bf16* __restrict__ out_relu_bf,
                      int ldc) {
    __shared__ __align__(16) bf16 sA[2][BM * BK];  // 16 KB double buffer
    __shared__ __align__(16) bf16 sB[BN * BK];     //  4 KB, [n][k]

    const int tid  = threadIdx.x;
    const int z    = blockIdx.z;
    const int m0   = blockIdx.x * BM;
    const int n0   = blockIdx.y * BN;
    const int wid  = tid >> 5;
    const int lane = tid & 31;
    const int half = lane >> 4;
    const int l15  = lane & 15;

    const float* Af = (const float*)Abase;
    const bf16*  Ab = (const bf16*)Abase;
    const float* Bf = (const float*)Bbase;
    const bf16*  Bb = (const bf16*)Bbase;

    // ---- per-thread base indices ----
    long aBaseIdx = 0, bBaseIdx;
    if constexpr (!ABF)
        aBaseIdx = (long)z * aBatch + (long)(m0 + (tid >> 3)) * lda + (tid & 7) * 4;
    if constexpr (BBF) bBaseIdx = (long)z * bBatch + (long)(tid >> 3) * ldb + n0 + (tid & 7) * 8;
    else               bBaseIdx = (long)z * bBatch + (long)(tid >> 4) * ldb + n0 + (tid & 15) * 4;

    const bf16* aTileBase = Ab + (long)z * aBatch + (long)m0 * lda;  // ABF/TDM path

    // staging registers
    float4 aF[4];
    float4 bF[2]; uint4 bH;

    auto tdmIssueA = [&](int buf, int k0) {
        tdm_load_tile_bf16((unsigned)(uintptr_t)(&sA[buf][0]),
                           aTileBase + k0,
                           (unsigned)lda, (unsigned)BM, (unsigned)lda,
                           (unsigned)BK, (unsigned)BM);
    };
    auto loadA = [&](int k0) {
#pragma unroll
        for (int i = 0; i < 4; ++i)
            aF[i] = *(const float4*)(Af + aBaseIdx + k0 + (long)i * 32 * lda);
    };
    auto storeA = [&](int buf) {
        uint2* sA2 = (uint2*)sA[buf];
#pragma unroll
        for (int i = 0; i < 4; ++i) sA2[tid + i * 256] = pack4_bf16(aF[i]);
    };
    auto loadB = [&](int k0) {
        if constexpr (BBF) {
            bH = *(const uint4*)(Bb + bBaseIdx + (long)k0 * ldb);
        } else {
#pragma unroll
            for (int i = 0; i < 2; ++i)
                bF[i] = *(const float4*)(Bf + bBaseIdx + (long)(k0 + i * 16) * ldb);
        }
    };
    auto storeB = [&]() {
        if constexpr (BBF) {
            const int kk = tid >> 3, no = (tid & 7) * 8;
            union { uint4 u; bf16 b[8]; } t; t.u = bH;
#pragma unroll
            for (int j = 0; j < 8; ++j) sB[(no + j) * BK + kk] = t.b[j];
        } else {
#pragma unroll
            for (int i = 0; i < 2; ++i) {
                const int kk = (tid >> 4) + i * 16, nq = tid & 15;
                sB[(nq * 4 + 0) * BK + kk] = (bf16)bF[i].x;
                sB[(nq * 4 + 1) * BK + kk] = (bf16)bF[i].y;
                sB[(nq * 4 + 2) * BK + kk] = (bf16)bF[i].z;
                sB[(nq * 4 + 3) * BK + kk] = (bf16)bF[i].w;
            }
        }
    };

    v8f c[4];
    const v8f zero = {0.f, 0.f, 0.f, 0.f, 0.f, 0.f, 0.f, 0.f};
#pragma unroll
    for (int j = 0; j < 4; ++j) c[j] = zero;

    const int nT = K / BK;

    // ---- pipeline preamble ----
    if constexpr (ABF) {
        if (wid == 0) tdmIssueA(0, 0);
    } else {
        loadA(0);
    }
    loadB(0);

    for (int t = 0; t < nT; ++t) {
        const int buf = t & 1;
        __syncthreads();                 // LDS free from previous compute
        storeB();
        if constexpr (!ABF) storeA(buf);
        if (t + 1 < nT) {
            if constexpr (ABF) {
                if (wid == 0) tdmIssueA(buf ^ 1, (t + 1) * BK);  // DMA next tile
            } else {
                loadA((t + 1) * BK);
            }
            loadB((t + 1) * BK);
            if constexpr (ABF) {
                if (wid == 0) __builtin_amdgcn_s_wait_tensorcnt(1);  // tile t landed
            }
        } else {
            if constexpr (ABF) {
                if (wid == 0) __builtin_amdgcn_s_wait_tensorcnt(0);
            }
        }
        __syncthreads();                 // sA[buf] + sB ready

        // A fragment (16-bit A 16x32 layout: lane<16 -> K{0..7,16..23},
        // lane>=16 -> K{8..15,24..31}; two ds_load_b128 per operand)
        union { v16bf v; uint4 u[2]; } af;
        const bf16* ap = &sA[buf][(wid * 16 + l15) * BK + half * 8];
        af.u[0] = *(const uint4*)(ap);
        af.u[1] = *(const uint4*)(ap + 16);

#pragma unroll
        for (int j = 0; j < 4; ++j) {
            union { v16bf v; uint4 u[2]; } bfrag;
            const bf16* bp = &sB[(j * 16 + l15) * BK + half * 8];
            bfrag.u[0] = *(const uint4*)(bp);
            bfrag.u[1] = *(const uint4*)(bp + 16);
            c[j] = __builtin_amdgcn_wmma_f32_16x16x32_bf16(
                false, af.v, false, bfrag.v, (short)0, c[j], false, false);
        }
    }

    // ---- epilogue (C layout: VGPR r -> M = half*8 + r, N = l15) ----
    const int baseRow = m0 + wid * 16 + half * 8;
#pragma unroll
    for (int j = 0; j < 4; ++j) {
        const int gcol = n0 + j * 16 + l15;
        float biasv = 0.f;
        if constexpr (EPI == EPI_BIAS_DUAL || EPI == EPI_BIAS_F32)
            biasv = bias[(long)z * biasBatch + gcol];
#pragma unroll
        for (int r = 0; r < 8; ++r) {
            const long ci = (long)(baseRow + r) * ldc + gcol;
            float v = c[j][r] + biasv;
            if constexpr (EPI == EPI_ADD_RELU) {
                v += (float)addend[(long)z * addBatch + ci];
                v = fmaxf(v, 0.f);
                ((bf16*)outPrim)[(long)z * outBatch + ci] = (bf16)v;
            } else if constexpr (EPI == EPI_PLAIN) {
                ((bf16*)outPrim)[(long)z * outBatch + ci] = (bf16)v;
            } else if constexpr (EPI == EPI_BIAS_DUAL) {
                ((bf16*)outPrim)[(long)z * outBatch + ci] = (bf16)v;
                out_relu_bf[(long)z * outBatch + ci] = (bf16)fmaxf(v, 0.f);
            } else { // EPI_BIAS_F32
                ((float*)outPrim)[(long)z * outBatch + ci] = v;
            }
        }
    }
}

// ---------------------------------------------------------------------------
// final[b,f] = mean over n of states[n,b,f]; vectorized 8 bf16 per thread.
// ---------------------------------------------------------------------------
__global__ __launch_bounds__(256)
void mean_nodes_kernel(const bf16* __restrict__ states,
                       bf16* __restrict__ finalv) {
    const long v8i = (long)blockIdx.x * blockDim.x + threadIdx.x;  // < 65536
    float s[8] = {0.f, 0.f, 0.f, 0.f, 0.f, 0.f, 0.f, 0.f};
    for (int n = 0; n < 128; ++n) {
        union { uint4 u; bf16 b[8]; } t;
        t.u = *(const uint4*)(states + (long)n * 524288 + v8i * 8);
#pragma unroll
        for (int e = 0; e < 8; ++e) s[e] += (float)t.b[e];
    }
    union { uint4 u; bf16 b[8]; } o;
#pragma unroll
    for (int e = 0; e < 8; ++e) o.b[e] = (bf16)(s[e] * (1.0f / 128.0f));
    *(uint4*)(finalv + v8i * 8) = o.u;
}

// ---------------------------------------------------------------------------
// Host-side orchestration
// ---------------------------------------------------------------------------
extern "C" void kernel_launch(void* const* d_in, const int* in_sizes, int n_in,
                              void* d_out, int out_size, void* d_ws, size_t ws_size,
                              hipStream_t stream) {
    (void)in_sizes; (void)n_in; (void)out_size; (void)ws_size;

    const float* x      = (const float*)d_in[0];   // (4096, 256)
    const float* nw     = (const float*)d_in[1];   // (128, 384, 128)
    const float* nbias  = (const float*)d_in[2];   // (128, 128)
    const float* att    = (const float*)d_in[3];   // (128, 128)
    const float* out_w  = (const float*)d_in[4];   // (128, 256)
    const float* out_b  = (const float*)d_in[5];   // (256,)
    float* out          = (float*)d_out;           // (4096, 256)

    // workspace layout
    char* ws = (char*)d_ws;
    float* conn   = (float*)(ws);                                  //  64 KB
    bf16*  xWb    = (bf16*)(ws + (65536));                         // 128 MB
    bf16*  states = (bf16*)(ws + (65536 + 134217728L));            // 128 MB
    bf16*  agg    = (bf16*)(ws + (65536 + 2 * 134217728L));        // 128 MB
    bf16*  finalb = (bf16*)(ws + (65536 + 3 * 134217728L));        //   1 MB

    const long BF   = 524288;   // B*F per node
    const long NWST = 49152;    // (D+F)*F per node in node_weight
    const float* W_x = nw + 128 * 128;  // rows F..F+D-1 of each node
    const float* W_s = nw;              // rows 0..F-1

    // 0) conn = softmax(attention)
    softmax_rows_kernel<<<1, 128, 0, stream>>>(att, conn);

    // 1) xW = x @ W_x[n] + bias[n]; states1 = relu(xW)  (iteration 1 fused)
    gemm_wmma_kernel<false, false, EPI_BIAS_DUAL><<<dim3(32, 2, 128), 256, 0, stream>>>(
        x, 0L, 256, W_x, NWST, 128, 256,
        nbias, 128L, nullptr, 0L,
        xWb, BF, states, 128);

    // iterations 2 and 3
    for (int it = 0; it < 2; ++it) {
        // 2) agg = conn @ states  (single GEMM: 128 x 524288, K=128)
        gemm_wmma_kernel<false, true, EPI_PLAIN><<<dim3(1, 8192, 1), 256, 0, stream>>>(
            conn, 0L, 128, states, 0L, (int)BF, 128,
            nullptr, 0L, nullptr, 0L,
            agg, 0L, nullptr, (int)BF);

        // 3) states = relu(agg @ W_s[n] + xW)   (A tile via TDM async DMA)
        gemm_wmma_kernel<true, false, EPI_ADD_RELU><<<dim3(32, 2, 128), 256, 0, stream>>>(
            agg, BF, 128, W_s, NWST, 128, 128,
            nullptr, 0L, xWb, BF,
            states, BF, nullptr, 128);
    }

    // 4) final = mean over nodes
    mean_nodes_kernel<<<256, 256, 0, stream>>>(states, finalb);

    // 5) out = final @ out_w + out_b  (f32 output; A tile via TDM)
    gemm_wmma_kernel<true, false, EPI_BIAS_F32><<<dim3(32, 4, 1), 256, 0, stream>>>(
        finalb, 0L, 128, out_w, 0L, 256, 128,
        out_b, 0L, nullptr, 0L,
        out, 0L, nullptr, 256);
}
